// StaticEncoderExport_25855703122018
// MI455X (gfx1250) — matrix operand order repaired and checked by reading, not compiled, not run on hardware
//
#include <hip/hip_runtime.h>
#include <hip/hip_bf16.h>
#include <math.h>

typedef __bf16 bf16;
typedef __bf16 v16bf __attribute__((ext_vector_type(16)));
typedef __bf16 v8bf  __attribute__((ext_vector_type(8)));
typedef __bf16 v4bf  __attribute__((ext_vector_type(4)));
typedef float  v8f   __attribute__((ext_vector_type(8)));
typedef float  v4f   __attribute__((ext_vector_type(4)));
typedef int    v4i   __attribute__((ext_vector_type(4)));

constexpr int Bc = 4, Tc = 4096, Dc = 1024, Hc = 16, HDc = 64, Lc = 6,
              Wc = 512, DFFc = 4096, DPc = 2048, TPCc = 512;
constexpr int Mc = Bc * Tc;   // 16384 token rows

// -------- CDNA5 async global->LDS staging (guarded; falls back to b128 copy) --
#if defined(__gfx1250__) && __has_builtin(__builtin_amdgcn_global_load_async_to_lds_b128) && __has_builtin(__builtin_amdgcn_s_wait_asynccnt)
#define ASYNC_LDS 1
#else
#define ASYNC_LDS 0
#endif

__device__ __forceinline__ void cp16(bf16* dst_lds, const bf16* src_g) {
#if ASYNC_LDS
  __builtin_amdgcn_global_load_async_to_lds_b128(
      (__attribute__((address_space(1))) v4i*)(__attribute__((address_space(1))) void*)(void*)src_g,
      (__attribute__((address_space(3))) v4i*)(__attribute__((address_space(3))) void*)(void*)dst_lds,
      0, 0);
#else
  *(v8bf*)dst_lds = *(const v8bf*)src_g;
#endif
}
__device__ __forceinline__ void cp_wait() {
#if ASYNC_LDS
  __builtin_amdgcn_s_wait_asynccnt(0);
#endif
}

// -------- WMMA fragment gathers, fully vectorized (2x ds_load_b128 each) -----

// A 16x32 bf16: lane&15 = row; per-lane K = khi+{0..7} and khi+16+{0..7}, khi = 8*(lane>>4)
__device__ __forceinline__ v16bf frag_a(const bf16* lds, int ld, int m0, int koff) {
  int lane = threadIdx.x & 31;
  const bf16* p = lds + (m0 + (lane & 15)) * ld + koff + ((lane >> 4) ? 8 : 0);
  v8bf lo = *(const v8bf*)p;
  v8bf hi = *(const v8bf*)(p + 16);
  return __builtin_shufflevector(lo, hi, 0, 1, 2, 3, 4, 5, 6, 7,
                                 8, 9, 10, 11, 12, 13, 14, 15);
}

// B 32x16 bf16 from LDS tile stored as [n][k]: lane&15 = N; per-lane K = khi+{0..7} and khi+16+{0..7}, khi = 8*(lane>>4)
__device__ __forceinline__ v16bf frag_bk(const bf16* lds, int ld, int n0, int koff) {
  int lane = threadIdx.x & 31;
  const bf16* p = lds + (n0 + (lane & 15)) * ld + koff + ((lane >> 4) ? 8 : 0);
  v8bf lo = *(const v8bf*)p;
  v8bf hi = *(const v8bf*)(p + 16);
  return __builtin_shufflevector(lo, hi, 0, 1, 2, 3, 4, 5, 6, 7,
                                 8, 9, 10, 11, 12, 13, 14, 15);
}

// ---------------- elementwise / LN kernels ----------------------------------

__global__ void enc_cvt(const float* __restrict__ src, bf16* __restrict__ dst) {
  long long i4 = ((long long)blockIdx.x * blockDim.x + threadIdx.x) * 4;
  v4f v = *(const v4f*)(src + i4);
  v4bf o;
#pragma unroll
  for (int j = 0; j < 4; ++j) o[j] = (bf16)v[j];
  *(v4bf*)(dst + i4) = o;
}

__global__ void enc_embed(const float* __restrict__ inp, const float* __restrict__ pos,
                          const unsigned char* __restrict__ tm, float* __restrict__ x) {
  long long i4 = ((long long)blockIdx.x * blockDim.x + threadIdx.x) * 4;  // < Mc*Dc
  int d   = (int)(i4 & (Dc - 1));
  int row = (int)(i4 >> 10);
  int t   = row & (Tc - 1);
  float m = tm[row] ? 1.f : 0.f;
  v4f a = *(const v4f*)(inp + i4);
  v4f p = *(const v4f*)(pos + (t & (TPCc - 1)) * Dc + d);
  v4f o;
#pragma unroll
  for (int j = 0; j < 4; ++j) o[j] = (a[j] + p[j]) * m;
  *(v4f*)(x + i4) = o;
}

__global__ void enc_ln(const float* __restrict__ x, const float* __restrict__ w,
                       const float* __restrict__ b, bf16* __restrict__ out) {
  __shared__ float rs[256], rq[256];
  int row = blockIdx.x, tid = threadIdx.x;
  const float* xr = x + (long long)row * Dc;
  v4f v = *(const v4f*)(xr + tid * 4);
  float s = v[0] + v[1] + v[2] + v[3];
  float q = v[0] * v[0] + v[1] * v[1] + v[2] * v[2] + v[3] * v[3];
  rs[tid] = s; rq[tid] = q; __syncthreads();
  for (int off = 128; off; off >>= 1) {
    if (tid < off) { rs[tid] += rs[tid + off]; rq[tid] += rq[tid + off]; }
    __syncthreads();
  }
  float mu  = rs[0] * (1.f / Dc);
  float var = rq[0] * (1.f / Dc) - mu * mu;
  float inv = rsqrtf(var + 1e-5f);
  v4f wv = *(const v4f*)(w + tid * 4);
  v4f bv = *(const v4f*)(b + tid * 4);
  v4bf o;
#pragma unroll
  for (int j = 0; j < 4; ++j) o[j] = (bf16)(((v[j] - mu) * inv) * wv[j] + bv[j]);
  *(v4bf*)(out + (long long)row * Dc + tid * 4) = o;
}

// ---------------- WMMA GEMM: Y = A[M,K](bf16) @ W[N,K]^T(bf16) + bias --------
// 128x64 block tile, 4 waves, each wave 32x64 (8 accumulators).
// Double-buffered async LDS staging: stage k+1 overlaps WMMA on tile k.
// epi: 0 = bf16 bias | 1 = bf16 bias+GELU | 2 = x += (acc+b)*mask (f32)
//      3 = out = (acc+b)*mask (f32) | 4 = x = (x+acc+b)*mask (f32)
//      5 = bf16 bias scattered into vT[win][head][hd][w]
__global__ void enc_gemm(const bf16* __restrict__ A, const bf16* __restrict__ Wb,
                         const float* __restrict__ bias, int N, int K,
                         bf16* __restrict__ Yb, float* __restrict__ Yf,
                         const unsigned char* __restrict__ tm, int epi) {
  constexpr int LDA = 40, LDB = 40;
  __shared__ __attribute__((aligned(16))) bf16 lA[2][128 * LDA];
  __shared__ __attribute__((aligned(16))) bf16 lB[2][64 * LDB];   // [n][k]
  int tid = threadIdx.x;
  int lane = tid & 31, wv = tid >> 5;
  int bm = blockIdx.x * 128, bn = blockIdx.y * 64;
  v8f c0[4] = {}, c1[4] = {};
  int nk = K >> 5;

  auto stage = [&](int kk, int buf) {
    int k0 = kk << 5;
#pragma unroll
    for (int i = 0; i < 4; ++i) {           // A: 128x32 bf16 = 512 chunks of 8
      int c = i * 128 + tid;
      int r = c >> 2, c8 = (c & 3) * 8;
      cp16(&lA[buf][r * LDA + c8], &A[(long long)(bm + r) * K + k0 + c8]);
    }
#pragma unroll
    for (int i = 0; i < 2; ++i) {           // B: 64x32 bf16 = 256 chunks of 8
      int c = i * 128 + tid;
      int n = c >> 2, k8 = (c & 3) * 8;
      cp16(&lB[buf][n * LDB + k8], &Wb[(long long)(bn + n) * K + k0 + k8]);
    }
  };

  stage(0, 0);
  for (int kk = 0; kk < nk; ++kk) {
    int cur = kk & 1;
    cp_wait();
    __syncthreads();
    if (kk + 1 < nk) stage(kk + 1, cur ^ 1);  // async under the WMMAs
    v16bf a0 = frag_a(lA[cur], LDA, wv * 32, 0);
    v16bf a1 = frag_a(lA[cur], LDA, wv * 32 + 16, 0);
#pragma unroll
    for (int j = 0; j < 4; ++j) {
      v16bf bf_ = frag_bk(lB[cur], LDB, j * 16, 0);
      c0[j] = __builtin_amdgcn_wmma_f32_16x16x32_bf16(false, a0, false, bf_,
                                                      (short)0, c0[j], false, false);
      c1[j] = __builtin_amdgcn_wmma_f32_16x16x32_bf16(false, a1, false, bf_,
                                                      (short)0, c1[j], false, false);
    }
    __syncthreads();
  }
  int g = lane >> 4, ln15 = lane & 15;
#pragma unroll
  for (int sub = 0; sub < 2; ++sub) {
    v8f* cc = sub ? c1 : c0;
#pragma unroll
    for (int j = 0; j < 4; ++j) {
#pragma unroll
      for (int r = 0; r < 8; ++r) {
        int gm = bm + wv * 32 + sub * 16 + r + 8 * g;
        int gn = bn + j * 16 + ln15;
        float vv = cc[j][r] + bias[gn];
        long long o = (long long)gm * N + gn;
        if (epi == 0) {
          Yb[o] = (bf16)vv;
        } else if (epi == 1) {
          Yb[o] = (bf16)(0.5f * vv * (1.f + erff(vv * 0.70710678f)));
        } else if (epi == 2) {
          float m = tm[gm] ? 1.f : 0.f;
          Yf[o] += vv * m;
        } else if (epi == 3) {
          float m = tm[gm] ? 1.f : 0.f;
          Yf[o] = vv * m;
        } else if (epi == 4) {
          float m = tm[gm] ? 1.f : 0.f;
          Yf[o] = (Yf[o] + vv) * m;
        } else {  // epi == 5: scatter into vT[(win*H+head)*HD+hd][w]
          int win = gm >> 9, w = gm & (Wc - 1);
          int head = gn >> 6, hd = gn & (HDc - 1);
          Yb[(((long long)(win * Hc + head) * HDc + hd) << 9) + w] = (bf16)vv;
        }
      }
    }
  }
}

// ---------------- flash attention over one (window, head, 64-row q tile) ----
__global__ void enc_attn(const bf16* __restrict__ q, const bf16* __restrict__ k,
                         const bf16* __restrict__ vT, const unsigned char* __restrict__ tm,
                         bf16* __restrict__ o) {
  constexpr int LDT = 72;
  __shared__ __attribute__((aligned(16))) bf16 Qt[64 * LDT];      // [qrow][hd]
  __shared__ __attribute__((aligned(16))) bf16 Kt[2][64 * LDT];   // [key][hd]  ([n][k] for S)
  __shared__ __attribute__((aligned(16))) bf16 Vt[2][64 * LDT];   // [hd][key]  ([n][k] for PV)
  __shared__ __attribute__((aligned(16))) bf16 Pt[64 * LDT];      // [qrow][key]
  __shared__ float km[2][64];
  int tid = threadIdx.x, lane = tid & 31, wv = tid >> 5;
  int qt = blockIdx.x, head = blockIdx.y, win = blockIdx.z;
  long long base = (long long)win * Wc;
  int hcol = head * HDc;
  const bf16* vhead = vT + (long long)(win * Hc + head) * HDc * Wc;

  auto stageKV = [&](int kc, int buf) {
#pragma unroll
    for (int i = 0; i < 4; ++i) {
      int c = i * 128 + tid;
      int r = c >> 3, c8 = (c & 7) * 8;
      cp16(&Kt[buf][r * LDT + c8], &k[(base + kc * 64 + r) * Dc + hcol + c8]);
      cp16(&Vt[buf][r * LDT + c8], &vhead[(long long)r * Wc + kc * 64 + c8]);
    }
    if (tid < 64) km[buf][tid] = tm[base + kc * 64 + tid] ? 0.f : -10000.f;
  };

#pragma unroll
  for (int i = 0; i < 4; ++i) {   // Q tile: 4096 elems, 512 chunks of 8
    int c = i * 128 + tid;
    int r = c >> 3, c8 = (c & 7) * 8;
    cp16(&Qt[r * LDT + c8], &q[(base + qt * 64 + r) * Dc + hcol + c8]);
  }
  stageKV(0, 0);

  float mrow[8], lsum[8];
#pragma unroll
  for (int r = 0; r < 8; ++r) { mrow[r] = -1e30f; lsum[r] = 0.f; }
  v8f oc[4] = {};
  int g = lane >> 4, ln15 = lane & 15;

  for (int kc = 0; kc < Wc / 64; ++kc) {
    int cur = kc & 1;
    cp_wait();
    __syncthreads();
    if (kc + 1 < Wc / 64) stageKV(kc + 1, cur ^ 1);   // async under compute

    v8f s[4] = {};
#pragma unroll
    for (int st = 0; st < 2; ++st) {
      v16bf af = frag_a(Qt, LDT, wv * 16, st * 32);
#pragma unroll
      for (int j = 0; j < 4; ++j) {
        v16bf bf_ = frag_bk(Kt[cur], LDT, j * 16, st * 32);
        s[j] = __builtin_amdgcn_wmma_f32_16x16x32_bf16(false, af, false, bf_,
                                                       (short)0, s[j], false, false);
      }
    }
    // online softmax per owned row (rows r + 8*g, replicated across 16-lane group)
#pragma unroll
    for (int r = 0; r < 8; ++r) {
      float sv[4], mx = -1e30f;
#pragma unroll
      for (int j = 0; j < 4; ++j) {
        sv[j] = s[j][r] * 0.125f + km[cur][j * 16 + ln15];
        mx = fmaxf(mx, sv[j]);
      }
#pragma unroll
      for (int d = 1; d < 16; d <<= 1) mx = fmaxf(mx, __shfl_xor(mx, d, 32));
      float mn = fmaxf(mrow[r], mx);
      float alpha = __expf(mrow[r] - mn);
      float ps = 0.f;
#pragma unroll
      for (int j = 0; j < 4; ++j) {
        float p = __expf(sv[j] - mn);
        ps += p;
        Pt[(wv * 16 + r + 8 * g) * LDT + j * 16 + ln15] = (bf16)p;
      }
#pragma unroll
      for (int d = 1; d < 16; d <<= 1) ps += __shfl_xor(ps, d, 32);
      lsum[r] = lsum[r] * alpha + ps;
      mrow[r] = mn;
#pragma unroll
      for (int j = 0; j < 4; ++j) oc[j][r] *= alpha;
    }
    __syncthreads();
    // O += P @ V  (B-frag from transposed V tile, vectorized)
#pragma unroll
    for (int st = 0; st < 2; ++st) {
      v16bf af = frag_a(Pt, LDT, wv * 16, st * 32);
#pragma unroll
      for (int j = 0; j < 4; ++j) {
        v16bf bf_ = frag_bk(Vt[cur], LDT, j * 16, st * 32);
        oc[j] = __builtin_amdgcn_wmma_f32_16x16x32_bf16(false, af, false, bf_,
                                                        (short)0, oc[j], false, false);
      }
    }
    __syncthreads();
  }
#pragma unroll
  for (int r = 0; r < 8; ++r) {
    float inv = 1.f / lsum[r];
    long long grow = base + qt * 64 + wv * 16 + r + 8 * g;
#pragma unroll
    for (int j = 0; j < 4; ++j)
      o[grow * Dc + hcol + j * 16 + ln15] = (bf16)(oc[j][r] * inv);
  }
}

// ---------------- host-side orchestration -----------------------------------

extern "C" void kernel_launch(void* const* d_in, const int* in_sizes, int n_in,
                              void* d_out, int out_size, void* d_ws, size_t ws_size,
                              hipStream_t stream) {
  (void)in_sizes; (void)n_in; (void)out_size; (void)ws_size;
  const float* inp = (const float*)d_in[0];
  const unsigned char* tm = (const unsigned char*)d_in[1];
  const float* pos = (const float*)d_in[2];
  const float* Wq = (const float*)d_in[3];  const float* bq = (const float*)d_in[4];
  const float* Wk = (const float*)d_in[5];  const float* bk = (const float*)d_in[6];
  const float* Wv = (const float*)d_in[7];  const float* bv = (const float*)d_in[8];
  const float* Wo = (const float*)d_in[9];  const float* bo = (const float*)d_in[10];
  const float* ln1w = (const float*)d_in[11]; const float* ln1b = (const float*)d_in[12];
  const float* ln2w = (const float*)d_in[13]; const float* ln2b = (const float*)d_in[14];
  const float* fc1w = (const float*)d_in[15]; const float* fc1b = (const float*)d_in[16];
  const float* fc2w = (const float*)d_in[17]; const float* fc2b = (const float*)d_in[18];
  const float* lnpw = (const float*)d_in[19]; const float* lnpb = (const float*)d_in[20];
  const float* p1w = (const float*)d_in[21];  const float* p1b = (const float*)d_in[22];
  const float* p2w = (const float*)d_in[23];  const float* p2b = (const float*)d_in[24];

  char* ws = (char*)d_ws;
  float* x  = (float*)ws;                                   // 64 MiB fp32 residual
  bf16* xn  = (bf16*)(ws + ((size_t)64  << 20));            // 32 MiB
  bf16* qb  = (bf16*)(ws + ((size_t)96  << 20));            // 32 MiB
  bf16* kb  = (bf16*)(ws + ((size_t)128 << 20));            // 32 MiB
  bf16* vTb = (bf16*)(ws + ((size_t)160 << 20));            // 32 MiB (V transposed/head)
  bf16* ob  = (bf16*)(ws + ((size_t)192 << 20));            // 32 MiB
  bf16* hb  = (bf16*)(ws + ((size_t)224 << 20));            // 128 MiB (FFN hidden / head)
  bf16* wbf = (bf16*)(ws + ((size_t)352 << 20));            // 156 MiB converted weights

  const long long szQKVO = (long long)Lc * Dc * Dc;         // 6291456
  const long long szFC   = (long long)Lc * DFFc * Dc;       // 25165824
  bf16* wq  = wbf;
  bf16* wk  = wq  + szQKVO;
  bf16* wv  = wk  + szQKVO;
  bf16* wo  = wv  + szQKVO;
  bf16* wf1 = wo  + szQKVO;
  bf16* wf2 = wf1 + szFC;
  bf16* wp1 = wf2 + szFC;                                   // DPc*Dc  = 2097152
  bf16* wp2 = wp1 + (long long)DPc * Dc;                    // DPc*DPc = 4194304

  // one-time (per launch) fp32 -> bf16 weight conversion
  enc_cvt<<<(int)(szQKVO / 1024), 256, 0, stream>>>(Wq, wq);
  enc_cvt<<<(int)(szQKVO / 1024), 256, 0, stream>>>(Wk, wk);
  enc_cvt<<<(int)(szQKVO / 1024), 256, 0, stream>>>(Wv, wv);
  enc_cvt<<<(int)(szQKVO / 1024), 256, 0, stream>>>(Wo, wo);
  enc_cvt<<<(int)(szFC / 1024), 256, 0, stream>>>(fc1w, wf1);
  enc_cvt<<<(int)(szFC / 1024), 256, 0, stream>>>(fc2w, wf2);
  enc_cvt<<<(DPc * Dc) / 1024, 256, 0, stream>>>(p1w, wp1);
  enc_cvt<<<(DPc * DPc) / 1024, 256, 0, stream>>>(p2w, wp2);

  enc_embed<<<(Mc * Dc / 4) / 256, 256, 0, stream>>>(inp, pos, tm, x);

  dim3 gD(Mc / 128, Dc / 64);
  for (int l = 0; l < Lc; ++l) {
    enc_ln<<<Mc, 256, 0, stream>>>(x, ln1w + l * Dc, ln1b + l * Dc, xn);
    enc_gemm<<<gD, 128, 0, stream>>>(xn, wq + (long long)l * Dc * Dc, bq + l * Dc,
                                     Dc, Dc, qb, nullptr, tm, 0);
    enc_gemm<<<gD, 128, 0, stream>>>(xn, wk + (long long)l * Dc * Dc, bk + l * Dc,
                                     Dc, Dc, kb, nullptr, tm, 0);
    enc_gemm<<<gD, 128, 0, stream>>>(xn, wv + (long long)l * Dc * Dc, bv + l * Dc,
                                     Dc, Dc, vTb, nullptr, tm, 5);
    enc_attn<<<dim3(Wc / 64, Hc, Bc * (Tc / Wc)), 128, 0, stream>>>(qb, kb, vTb, tm, ob);
    enc_gemm<<<gD, 128, 0, stream>>>(ob, wo + (long long)l * Dc * Dc, bo + l * Dc,
                                     Dc, Dc, nullptr, x, tm, 2);
    enc_ln<<<Mc, 256, 0, stream>>>(x, ln2w + l * Dc, ln2b + l * Dc, xn);
    enc_gemm<<<dim3(Mc / 128, DFFc / 64), 128, 0, stream>>>(
        xn, wf1 + (long long)l * DFFc * Dc, fc1b + l * DFFc, DFFc, Dc, hb, nullptr, tm, 1);
    enc_gemm<<<gD, 128, 0, stream>>>(hb, wf2 + (long long)l * Dc * DFFc, fc2b + l * Dc,
                                     Dc, DFFc, nullptr, x, tm, 4);
  }

  enc_ln<<<Mc, 256, 0, stream>>>(x, lnpw, lnpb, xn);
  enc_gemm<<<dim3(Mc / 128, DPc / 64), 128, 0, stream>>>(xn, wp1, p1b, DPc, Dc,
                                                         hb, nullptr, tm, 1);
  enc_gemm<<<dim3(Mc / 128, DPc / 64), 128, 0, stream>>>(hb, wp2, p2b, DPc, DPc,
                                                         nullptr, (float*)d_out, tm, 3);
}
